// EssenceNetClassifier_5007931867322
// MI455X (gfx1250) — matrix operand order repaired and checked by reading, not compiled, not run on hardware
//
#include <hip/hip_runtime.h>
#include <hip/hip_bf16.h>
#include <math.h>

typedef __attribute__((ext_vector_type(16))) __bf16 v16bf;
typedef __attribute__((ext_vector_type(8)))  float  v8f;

union ABv { unsigned int u[8]; v16bf v; };

__device__ __forceinline__ unsigned short f2bf(float f){
  unsigned int u = __float_as_uint(f);
  unsigned int r = u + 0x7FFFu + ((u >> 16) & 1u);
  return (unsigned short)(r >> 16);
}
__device__ __forceinline__ unsigned int ordkey(float f){
  unsigned int u = __float_as_uint(f);
  return (u & 0x80000000u) ? ~u : (u | 0x80000000u);
}
__device__ __forceinline__ float ordinv(unsigned int u){
  unsigned int r = (u & 0x80000000u) ? (u & 0x7FFFFFFFu) : ~u;
  return __uint_as_float(r);
}
// bf16 activation buffers use K-pair interleaved layout:
// element (c, p) -> ushort index ((c>>1)*P + p)*2 + (c&1)   (== dword [c>>1][p])
__device__ __forceinline__ size_t ilv(int c, int p, int P){
  return ((size_t)(c >> 1) * P + p) * 2 + (c & 1);
}

// ---------------- utility kernels ----------------
__global__ void zero_u32(unsigned int* p, long n){
  long i = (long)blockIdx.x*blockDim.x + threadIdx.x;
  if (i < n) p[i] = 0u;
}

__global__ void bn_prep(const float* __restrict__ g, const float* __restrict__ b,
                        const float* __restrict__ m, const float* __restrict__ v,
                        float* __restrict__ scale, float* __restrict__ bias, int C){
  int c = blockIdx.x*blockDim.x + threadIdx.x;
  if (c >= C) return;
  float s = g[c] * rsqrtf(v[c] + 1e-5f);
  scale[c] = s;
  bias[c]  = b[c] - m[c]*s;
}

// f32 [OC, IC] -> bf16 [OC, Kp] row-major, zero-padded in K (A operand)
__global__ void stage_w(const float* __restrict__ w, unsigned short* __restrict__ dst,
                        int OC, int IC, int Kp){
  long i = (long)blockIdx.x*blockDim.x + threadIdx.x;
  long total = (long)OC*Kp;
  if (i >= total) return;
  int o = (int)(i / Kp), k = (int)(i % Kp);
  dst[i] = (k < IC) ? f2bf(w[(size_t)o*IC + k]) : (unsigned short)0;
}

__global__ void gray_kernel(const float* __restrict__ x, float* __restrict__ gray){
  int i = blockIdx.x*blockDim.x + threadIdx.x;
  if (i >= 2*65536) return;
  int b = i / 65536, pos = i % 65536;
  const float* xb = x + (size_t)b*3*65536;
  gray[i] = 0.299f*xb[pos] + 0.587f*xb[65536 + pos] + 0.114f*xb[2*65536 + pos];
}

// backbone conv1 (1 input channel, big kernel) + BN + SiLU -> bf16 interleaved [midp, P]
__global__ void conv1_bn_silu(const float* __restrict__ gray, const float* __restrict__ W,
                              const float* __restrict__ scale, const float* __restrict__ bias,
                              unsigned short* __restrict__ out,
                              int mid, int midp, int H, int Wd, int ks, int s, int pad){
  int HW = H*Wd; int P = 2*HW;
  long idx = (long)blockIdx.x*blockDim.x + threadIdx.x;
  long total = (long)midp*P;
  if (idx >= total) return;
  int c = (int)(idx / P), p = (int)(idx % P);
  if (c >= mid){ out[ilv(c, p, P)] = 0; return; }
  int b = p / HW, pos = p % HW;
  int oy = pos / Wd, ox = pos % Wd;
  const float* g = gray + (size_t)b*65536;
  const float* w = W + (size_t)c*ks*ks;
  float acc = 0.f;
  int iy0 = oy*s - pad, ix0 = ox*s - pad;
  for (int ky = 0; ky < ks; ++ky){
    int iy = iy0 + ky; if (iy < 0 || iy >= 256) continue;
    for (int kx = 0; kx < ks; ++kx){
      int ix = ix0 + kx; if (ix < 0 || ix >= 256) continue;
      acc += g[iy*256 + ix] * w[ky*ks + kx];
    }
  }
  float y = acc * scale[c] + bias[c];
  y = y / (1.f + expf(-y));
  out[ilv(c, p, P)] = f2bf(y);
}

// area pool RGB input to H x W -> channels [0..2] of interleaved concat buffer
__global__ void area_pool(const float* __restrict__ x, unsigned short* __restrict__ out,
                          int H, int Wd){
  int HW = H*Wd; int P = 2*HW;
  int idx = blockIdx.x*blockDim.x + threadIdx.x;
  if (idx >= 3*P) return;
  int c = idx / P, p = idx % P;
  int b = p / HW, pos = p % HW;
  int oy = pos / Wd, ox = pos % Wd;
  int fy = 256 / H, fx = 256 / Wd;
  const float* src = x + ((size_t)b*3 + c)*65536;
  float s = 0.f;
  for (int dy = 0; dy < fy; ++dy)
    for (int dx = 0; dx < fx; ++dx)
      s += src[(oy*fy+dy)*256 + (ox*fx+dx)];
  out[ilv(c, p, P)] = f2bf(s / (float)(fy*fx));
}

// nearest upsample bf16 interleaved [C, Pp] -> dst channels [rowOff, rowOff+C) interleaved [*, P]
__global__ void upsample_bf16(const unsigned short* __restrict__ vv, unsigned short* __restrict__ dst,
                              int C, int rowOff, int H, int Wd, int Hp, int Wp){
  int HW = H*Wd, HWp = Hp*Wp; int P = 2*HW, Pp = 2*HWp;
  long idx = (long)blockIdx.x*blockDim.x + threadIdx.x;
  if (idx >= (long)C*P) return;
  int c = (int)(idx / P), p = (int)(idx % P);
  int b = p / HW, pos = p % HW;
  int y = pos / Wd, xq = pos % Wd;
  int ys = (y*Hp)/H, xs = (xq*Wp)/Wd;
  int psrc = b*HWp + ys*Wp + xs;
  dst[ilv(c + rowOff, p, P)] = vv[ilv(c, psrc, Pp)];
}

// ---------------- WMMA bf16 GEMM: C[M,P] = A[M,Kp] * B[Kp,P] + epilogue ----------------
// A row-major bf16 [M rows, Kp]; B pair-interleaved dwords [Kp/2][P]; Kp % 32 == 0.
// Wave computes a 16x64 tile: A fragment reused across 4 WMMAs per K-step.
// Addressing: FIXED kernarg SGPR base + 32-bit element offsets (scalar K-advance
// `kb` + lane offsets), matching the GVS saddr+voffset global-load pattern.
// All loads unconditional via index clamping (garbage rows/cols never stored).
// mode 0: y = silu(c*scale[m]+bias[m]).  mode 1: y = c + bias[m].
// Rows [M_real, M_store) written as zeros. Output bf16-interleaved (outB, channel m+rowOff)
// or f32 row-major (outF).
__global__ void gemm_wmma(const unsigned int* __restrict__ Aw,
                          const unsigned int* __restrict__ Bw,
                          unsigned short* __restrict__ outB,
                          float* __restrict__ outF,
                          const float* __restrict__ scale,
                          const float* __restrict__ bias,
                          int M_real, int M_store, int rowOff, int Kp, int P,
                          int mode, int tilesN, int totalTiles){
  int wid  = threadIdx.x >> 5;
  int lane = threadIdx.x & 31;
  int t = blockIdx.x * (blockDim.x >> 5) + wid;
  if (t >= totalTiles) return;
  int tm  = (t / tilesN) * 16;
  int tn0 = (t % tilesN) * 64;
  int h = lane >> 4;
  int l = lane & 15;
  int Kw = Kp >> 1;                                              // dword rows of B / A
  unsigned int uP = (unsigned int)P;

  int arow = tm + l; if (arow >= M_real) arow = M_real - 1;      // clamp: pad rows discarded
  // A lane offsets (32-bit): word = arow*Kw + h*4 + v (+8) + kk2(scalar)
  unsigned int aoff[8];
#pragma unroll
  for (int v = 0; v < 4; ++v){
    aoff[v]   = (unsigned int)(arow * Kw + h*4 + v);
    aoff[v+4] = aoff[v] + 8u;
  }
  // B lane offsets (32-bit): word = [kb + (v + (v>>2)*4)*P](scalar) + [h*4*P + n](lane)
  unsigned int jb[4];
#pragma unroll
  for (int j = 0; j < 4; ++j){
    int n = tn0 + 16*j + l;
    if (n >= P) n = P - 1;                                       // clamp: pad cols discarded
    jb[j] = (unsigned int)h*4u*uP + (unsigned int)n;
  }

  v8f acc0 = {}, acc1 = {}, acc2 = {}, acc3 = {};
  unsigned int kb = 0u;                                          // scalar: kk2 * P
  for (int kk2 = 0; kk2 < Kw; kk2 += 16, kb += 16u*uP){
    __builtin_prefetch(Aw + (unsigned int)(kk2 + 16) + aoff[0], 0, 1);
    __builtin_prefetch(Bw + kb + 16u*uP + jb[0], 0, 1);
    ABv a;
#pragma unroll
    for (int v = 0; v < 8; ++v)
      a.u[v] = Aw[aoff[v] + (unsigned int)kk2];                  // saddr + 32-bit voffset
    ABv b0, b1, b2, b3;
#pragma unroll
    for (int v = 0; v < 8; ++v){
      unsigned int bo = kb + (unsigned int)(v + (v >> 2) * 4) * uP; // scalar per v
      b0.u[v] = Bw[bo + jb[0]];
      b1.u[v] = Bw[bo + jb[1]];
      b2.u[v] = Bw[bo + jb[2]];
      b3.u[v] = Bw[bo + jb[3]];
    }
    acc0 = __builtin_amdgcn_wmma_f32_16x16x32_bf16(false, a.v, false, b0.v, (short)0, acc0, false, false);
    acc1 = __builtin_amdgcn_wmma_f32_16x16x32_bf16(false, a.v, false, b1.v, (short)0, acc1, false, false);
    acc2 = __builtin_amdgcn_wmma_f32_16x16x32_bf16(false, a.v, false, b2.v, (short)0, acc2, false, false);
    acc3 = __builtin_amdgcn_wmma_f32_16x16x32_bf16(false, a.v, false, b3.v, (short)0, acc3, false, false);
  }

  v8f acc[4] = {acc0, acc1, acc2, acc3};
#pragma unroll
  for (int j = 0; j < 4; ++j){
    int n = tn0 + 16*j + l;
    if (n >= P) continue;
#pragma unroll
    for (int r = 0; r < 8; ++r){
      int m = tm + r + 8*h;                                      // C layout: VGPR r, half h
      if (m >= M_store) continue;
      float y = 0.f;
      if (m < M_real){
        float c = acc[j][r];
        if (mode == 0){
          y = c * scale[m] + bias[m];
          y = y / (1.f + expf(-y));
        } else {
          y = c + bias[m];
        }
      }
      if (outF) outF[(size_t)m*P + n] = y;
      else      outB[ilv(m + rowOff, n, P)] = f2bf(y);
    }
  }
}

// ---------------- instance norm stats / fused norm-add ----------------
__global__ void reduce_ms(const float* __restrict__ L, float* __restrict__ stats, int C, int HW, int P){
  long total = (long)C*P;
  float a0=0,q0=0,a1=0,q1=0;
  for (long i = (long)blockIdx.x*blockDim.x + threadIdx.x; i < total; i += (long)gridDim.x*blockDim.x){
    int p = (int)(i % P);
    float v = L[i];
    if (p < HW){ a0 += v; q0 += v*v; } else { a1 += v; q1 += v*v; }
  }
  __shared__ float s[4][256];
  s[0][threadIdx.x]=a0; s[1][threadIdx.x]=q0; s[2][threadIdx.x]=a1; s[3][threadIdx.x]=q1;
  __syncthreads();
  for (int off = 128; off > 0; off >>= 1){
    if ((int)threadIdx.x < off)
      for (int j = 0; j < 4; ++j) s[j][threadIdx.x] += s[j][threadIdx.x + off];
    __syncthreads();
  }
  if (threadIdx.x == 0){
    atomicAdd(&stats[0], s[0][0]); atomicAdd(&stats[1], s[1][0]);
    atomicAdd(&stats[2], s[2][0]); atomicAdd(&stats[3], s[3][0]);
  }
}

// Ln = inst_norm(nearest(Lp)) + inst_norm(Li); stats[0..3]=prev sums, [4..7]=Li sums.
__global__ void norm_add(const float* __restrict__ Lp, const float* __restrict__ Li,
                         float* __restrict__ Ln, const float* __restrict__ st,
                         int C, int H, int Wd, int Hp, int Wp){
  int HWc = H*Wd, HWp = Hp*Wp;
  int P = 2*HWc, Pp = 2*HWp;
  long idx = (long)blockIdx.x*blockDim.x + threadIdx.x;
  if (idx >= (long)C*P) return;
  int c = (int)(idx / P), p = (int)(idx % P);
  int b = p / HWc, pos = p % HWc;
  int y = pos / Wd, xq = pos % Wd;
  int ys = (y*Hp)/H, xs = (xq*Wp)/Wd;

  float f2   = (float)HWc / (float)HWp;          // nearest-x2 replication factor
  float n_up = (float)C * (float)HWc;
  float sp = st[2*b], qp = st[2*b+1];
  float si = st[4 + 2*b], qi = st[5 + 2*b];
  float mean_p = sp / ((float)C * (float)HWp);   // == mean of the upsampled tensor
  float var_p  = (f2*qp - n_up*mean_p*mean_p) / (n_up - 1.f);
  float sd_p   = sqrtf(fmaxf(var_p, 0.f));
  float mean_i = si / n_up;
  float var_i  = (qi - n_up*mean_i*mean_i) / (n_up - 1.f);
  float sd_i   = sqrtf(fmaxf(var_i, 0.f));

  float pv = Lp[(size_t)c*Pp + (size_t)b*HWp + (size_t)ys*Wp + xs];
  Ln[idx] = (pv - mean_p)/(sd_p + 1e-5f) + (Li[idx] - mean_i)/(sd_i + 1e-5f);
}

// ---------------- final selection ----------------
__global__ void pos_argmax(const float* __restrict__ L, int* __restrict__ amax,
                           float* __restrict__ mval, int C, int P){
  int p = blockIdx.x*blockDim.x + threadIdx.x;
  if (p >= P) return;
  float best = L[p]; int bi = 0;
  for (int c = 1; c < C; ++c){
    float v = L[(size_t)c*P + p];
    if (v > best){ best = v; bi = c; }
  }
  amax[p] = bi; mval[p] = best;
}

__global__ void select_kernel(const float* __restrict__ L, const int* __restrict__ amax,
                              const float* __restrict__ mval, float* __restrict__ out,
                              int C, int HW){
  int b = blockIdx.x;
  int P = 2*HW, base = b*HW;
  int tid = threadIdx.x, nt = blockDim.x;
  __shared__ float sf[1024];
  __shared__ int   si[1024];

  // 1) max_pos = argmax_n mval (first occurrence)
  float bv = -INFINITY; int bi = 0x7FFFFFFF;
  for (int n = tid; n < HW; n += nt){
    float v = mval[base + n];
    if (v > bv || (v == bv && n < bi)){ bv = v; bi = n; }
  }
  sf[tid] = bv; si[tid] = bi; __syncthreads();
  for (int off = nt/2; off > 0; off >>= 1){
    if (tid < off){
      float v = sf[tid+off]; int j = si[tid+off];
      if (v > sf[tid] || (v == sf[tid] && j < si[tid])){ sf[tid] = v; si[tid] = j; }
    }
    __syncthreads();
  }
  int max_pos = si[0]; __syncthreads();

  // 2) class_id = argmax_c L[c, max_pos]
  bv = -INFINITY; bi = 0x7FFFFFFF;
  for (int c = tid; c < C; c += nt){
    float v = L[(size_t)c*P + base + max_pos];
    if (v > bv || (v == bv && c < bi)){ bv = v; bi = c; }
  }
  sf[tid] = bv; si[tid] = bi; __syncthreads();
  for (int off = nt/2; off > 0; off >>= 1){
    if (tid < off){
      float v = sf[tid+off]; int j = si[tid+off];
      if (v > sf[tid] || (v == sf[tid] && j < si[tid])){ sf[tid] = v; si[tid] = j; }
    }
    __syncthreads();
  }
  int class_id = si[0]; __syncthreads();

  // 3) n_valid
  int cnt = 0;
  for (int n = tid; n < HW; n += nt) cnt += (amax[base + n] == class_id) ? 1 : 0;
  si[tid] = cnt; __syncthreads();
  for (int off = nt/2; off > 0; off >>= 1){
    if (tid < off) si[tid] += si[tid+off];
    __syncthreads();
  }
  int n_valid = si[0]; __syncthreads();

  // 4) lower-median via 32-step radix select
  int k = (n_valid - 1) / 2;
  unsigned int prefix = 0u;
  for (int bit = 31; bit >= 0; --bit){
    unsigned int ph = prefix >> bit;
    int c_ = 0;
    for (int n = tid; n < HW; n += nt){
      unsigned int key;
      if (amax[base + n] == class_id) key = ordkey(L[(size_t)class_id*P + base + n]);
      else key = 0xFFFFFFFFu;
      if ((key >> bit) == ph) c_++;
    }
    si[tid] = c_; __syncthreads();
    for (int off = nt/2; off > 0; off >>= 1){
      if (tid < off) si[tid] += si[tid+off];
      __syncthreads();
    }
    int cnt0 = si[0]; __syncthreads();
    if (k >= cnt0){ k -= cnt0; prefix |= (1u << bit); }
  }
  float median = ordinv(prefix);

  // 5) idx = argmin_n |score - median| over mask (first occurrence)
  float bd = INFINITY; int bn = 0x7FFFFFFF;
  for (int n = tid; n < HW; n += nt){
    if (amax[base + n] == class_id){
      float d = fabsf(L[(size_t)class_id*P + base + n] - median);
      if (d < bd || (d == bd && n < bn)){ bd = d; bn = n; }
    }
  }
  sf[tid] = bd; si[tid] = bn; __syncthreads();
  for (int off = nt/2; off > 0; off >>= 1){
    if (tid < off){
      float v = sf[tid+off]; int j = si[tid+off];
      if (v < sf[tid] || (v == sf[tid] && j < si[tid])){ sf[tid] = v; si[tid] = j; }
    }
    __syncthreads();
  }
  int idx = si[0]; __syncthreads();

  // 6) emit full logit vector at idx
  for (int c = tid; c < C; c += nt)
    out[(size_t)b*C + c] = L[(size_t)c*P + base + idx];
}

// ---------------- host orchestration ----------------
static inline size_t align256(size_t x){ return (x + 255) & ~(size_t)255; }
static inline int pad32(int x){ return (x + 31) / 32 * 32; }
static inline int cdiv(long a, int b){ return (int)((a + b - 1) / b); }

extern "C" void kernel_launch(void* const* d_in, const int* in_sizes, int n_in,
                              void* d_out, int out_size, void* d_ws, size_t ws_size,
                              hipStream_t stream){
  (void)n_in; (void)out_size; (void)ws_size;
  static const int MID[8] = {2048,1024,512,256,128,64,32,16};
  static const int OUT[8] = {1024,512,256,128,64,32,16,8};
  static const int KS[8]  = {256,128,64,32,16,8,4,3};
  static const int ST[8]  = {256,128,64,32,16,8,4,2};
  static const int PD[8]  = {0,0,0,0,0,0,0,1};
  const int VCS = 1024, NCLS = 1000, HHID = 512;

  int Hh[8], Wh[8], inC[8], hidC[8];
  for (int i = 0; i < 8; ++i){
    Hh[i] = (256 + 2*PD[i] - KS[i]) / ST[i] + 1;
    Wh[i] = Hh[i];
    inC[i] = (i == 0) ? (OUT[0] + 3) : (OUT[i] + 3 + VCS);
    hidC[i] = inC[i] / 2;
  }

  // ---- map input leaves (insertion order vs jax-sorted pytree order) ----
  const float *x, *bb_w1[8], *bb_w2[8], *en_w1[8], *en_w2[8], *en_b2[8];
  const float *bb1g[8], *bb1b[8], *bb1m[8], *bb1v[8];
  const float *bb2g[8], *bb2b[8], *bb2m[8], *bb2v[8];
  const float *eng[8], *enb[8], *enm[8], *env[8];
  const float *hd_w1, *hd_w2, *hd_b2, *hdg, *hdb, *hdm, *hdv;
  {
    int ci = 0;
    auto nx = [&]() -> const float* { return (const float*)d_in[ci++]; };
    bool orderA = (in_sizes[0] == 2*3*256*256);
    if (orderA){
      x = nx();
      for (int i = 0; i < 8; ++i){
        bb_w1[i]=nx(); bb1g[i]=nx(); bb1b[i]=nx(); bb1m[i]=nx(); bb1v[i]=nx();
        bb_w2[i]=nx(); bb2g[i]=nx(); bb2b[i]=nx(); bb2m[i]=nx(); bb2v[i]=nx();
      }
      for (int i = 0; i < 8; ++i){
        en_w1[i]=nx(); eng[i]=nx(); enb[i]=nx(); enm[i]=nx(); env[i]=nx();
        en_w2[i]=nx(); en_b2[i]=nx();
      }
      hd_w1=nx(); hdg=nx(); hdb=nx(); hdm=nx(); hdv=nx(); hd_w2=nx(); hd_b2=nx();
    } else {
      for (int i = 0; i < 8; ++i){
        bb1b[i]=nx(); bb1g[i]=nx(); bb1m[i]=nx(); bb1v[i]=nx();
        bb2b[i]=nx(); bb2g[i]=nx(); bb2m[i]=nx(); bb2v[i]=nx();
        bb_w1[i]=nx(); bb_w2[i]=nx();
      }
      for (int i = 0; i < 8; ++i){
        en_b2[i]=nx(); enb[i]=nx(); eng[i]=nx(); enm[i]=nx(); env[i]=nx();
        en_w1[i]=nx(); en_w2[i]=nx();
      }
      hd_b2=nx(); hdb=nx(); hdg=nx(); hdm=nx(); hdv=nx(); hd_w1=nx(); hd_w2=nx();
      x = nx();
    }
  }

  // ---- workspace arena ----
  char* wsb = (char*)d_ws; size_t off = 0;
  auto alloc = [&](size_t bytes) -> void* { void* p = wsb + off; off = align256(off + bytes); return p; };

  float* gray = (float*)alloc(2*65536*sizeof(float));
  unsigned short *w2s[8], *ew1s[8], *ew2s[8];
  float *bn1s[8], *bn1bi[8], *bn2s[8], *bn2bi[8], *ebns[8], *ebnbi[8];
  for (int i = 0; i < 8; ++i){
    w2s[i]  = (unsigned short*)alloc((size_t)OUT[i]*pad32(MID[i])*2);
    ew1s[i] = (unsigned short*)alloc((size_t)hidC[i]*pad32(inC[i])*2);
    ew2s[i] = (unsigned short*)alloc((size_t)VCS*pad32(hidC[i])*2);
    bn1s[i]=(float*)alloc(MID[i]*4);  bn1bi[i]=(float*)alloc(MID[i]*4);
    bn2s[i]=(float*)alloc(OUT[i]*4);  bn2bi[i]=(float*)alloc(OUT[i]*4);
    ebns[i]=(float*)alloc(hidC[i]*4); ebnbi[i]=(float*)alloc(hidC[i]*4);
  }
  unsigned short* hw1s = (unsigned short*)alloc((size_t)HHID*VCS*2);
  unsigned short* hw2s = (unsigned short*)alloc((size_t)NCLS*HHID*2);
  float* hbns  = (float*)alloc(HHID*4);
  float* hbnbi = (float*)alloc(HHID*4);

  int Pmax = 2*Hh[7]*Wh[7];
  size_t midMax = 0, xiMax = 0, hidMax = 0;
  for (int i = 0; i < 8; ++i){
    size_t P = (size_t)2*Hh[i]*Wh[i];
    midMax = midMax > (size_t)pad32(MID[i])*P ? midMax : (size_t)pad32(MID[i])*P;
    xiMax  = xiMax  > (size_t)pad32(inC[i])*P ? xiMax  : (size_t)pad32(inC[i])*P;
    hidMax = hidMax > (size_t)pad32(hidC[i])*P ? hidMax : (size_t)pad32(hidC[i])*P;
  }
  unsigned short* Hm = (unsigned short*)alloc(midMax*2);
  unsigned short* Xi = (unsigned short*)alloc(xiMax*2);
  unsigned short* Hd = (unsigned short*)alloc(hidMax*2);
  unsigned short* Vv = (unsigned short*)alloc((size_t)VCS*Pmax*2);
  unsigned short* Hq = (unsigned short*)alloc((size_t)HHID*Pmax*2);
  float* Lb[3];
  for (int j = 0; j < 3; ++j) Lb[j] = (float*)alloc((size_t)NCLS*Pmax*4);
  float* stats = (float*)alloc(8*4);
  int*   amax  = (int*)alloc((size_t)Pmax*4);
  float* mval  = (float*)alloc((size_t)Pmax*4);

  // ---- staging ----
  auto launch_stage = [&](const float* w, unsigned short* dst, int OC, int IC, int Kp){
    long n = (long)OC*Kp;
    stage_w<<<cdiv(n,256), 256, 0, stream>>>(w, dst, OC, IC, Kp);
  };
  auto launch_bn = [&](const float* g, const float* b, const float* m, const float* v,
                       float* sc, float* bi, int C){
    bn_prep<<<cdiv(C,256), 256, 0, stream>>>(g, b, m, v, sc, bi, C);
  };
  for (int i = 0; i < 8; ++i){
    launch_stage(bb_w2[i], w2s[i], OUT[i], MID[i], pad32(MID[i]));
    launch_stage(en_w1[i], ew1s[i], hidC[i], inC[i], pad32(inC[i]));
    launch_stage(en_w2[i], ew2s[i], VCS, hidC[i], pad32(hidC[i]));
    launch_bn(bb1g[i], bb1b[i], bb1m[i], bb1v[i], bn1s[i], bn1bi[i], MID[i]);
    launch_bn(bb2g[i], bb2b[i], bb2m[i], bb2v[i], bn2s[i], bn2bi[i], OUT[i]);
    launch_bn(eng[i],  enb[i],  enm[i],  env[i],  ebns[i], ebnbi[i], hidC[i]);
  }
  launch_stage(hd_w1, hw1s, HHID, VCS, VCS);
  launch_stage(hd_w2, hw2s, NCLS, HHID, HHID);
  launch_bn(hdg, hdb, hdm, hdv, hbns, hbnbi, HHID);

  gray_kernel<<<cdiv(2*65536,256), 256, 0, stream>>>(x, gray);

  auto gemm = [&](const unsigned short* A, const unsigned short* B,
                  unsigned short* oB, float* oF, const float* sc, const float* bi,
                  int Mr, int Ms, int rowOff, int Kp, int P, int mode){
    int tilesM = (Ms + 15)/16, tilesN = (P + 63)/64;
    int total = tilesM * tilesN;
    gemm_wmma<<<(total + 3)/4, 128, 0, stream>>>((const unsigned int*)A, (const unsigned int*)B,
                                                 oB, oF, sc, bi,
                                                 Mr, Ms, rowOff, Kp, P, mode, tilesN, total);
  };

  // ---- multi-scale pipeline ----
  int prev = 0;
  for (int i = 0; i < 8; ++i){
    int H = Hh[i], W = Wh[i], HW = H*W, P = 2*HW;
    int midp = pad32(MID[i]), Cp = pad32(inC[i]), hidp = pad32(hidC[i]);

    // backbone conv1 (patch conv) + BN + SiLU
    {
      long n = (long)midp*P;
      conv1_bn_silu<<<cdiv(n,256), 256, 0, stream>>>(gray, bb_w1[i], bn1s[i], bn1bi[i],
                                                     Hm, MID[i], midp, H, W, KS[i], ST[i], PD[i]);
    }
    // zero + build concat buffer Xi = [rc(3) | kf(out) | vv_up(1024) | pad]
    {
      long n = ((long)Cp*P)/2;
      zero_u32<<<cdiv(n,256), 256, 0, stream>>>((unsigned int*)Xi, n);
    }
    area_pool<<<cdiv((long)3*P,256), 256, 0, stream>>>(x, Xi, H, W);
    gemm(w2s[i], Hm, Xi, nullptr, bn2s[i], bn2bi[i], OUT[i], OUT[i], /*rowOff=*/3, midp, P, 0);
    if (i > 0){
      long n = (long)VCS*P;
      upsample_bf16<<<cdiv(n,256), 256, 0, stream>>>(Vv, Xi, VCS, 3 + OUT[i], H, W, Hh[i-1], Wh[i-1]);
    }
    // encoder
    gemm(ew1s[i], Xi, Hd, nullptr, ebns[i], ebnbi[i], hidC[i], hidp, 0, Cp, P, 0);
    gemm(ew2s[i], Hd, Vv, nullptr, nullptr, en_b2[i], VCS, VCS, 0, hidp, P, 1);
    // head
    gemm(hw1s, Vv, Hq, nullptr, hbns, hbnbi, HHID, HHID, 0, VCS, P, 0);
    float* Li = (i == 0) ? Lb[0] : Lb[(prev+1)%3];
    gemm(hw2s, Hq, nullptr, Li, nullptr, hd_b2, NCLS, NCLS, 0, HHID, P, 1);

    if (i > 0){
      int Hp = Hh[i-1], Wp = Wh[i-1], HWp = Hp*Wp, Pp = 2*HWp;
      zero_u32<<<1, 8, 0, stream>>>((unsigned int*)stats, 8);
      {
        long tp = (long)NCLS*Pp;
        int bl = cdiv(tp, 256); if (bl > 2048) bl = 2048;
        reduce_ms<<<bl, 256, 0, stream>>>(Lb[prev], stats, NCLS, HWp, Pp);
        long tc = (long)NCLS*P;
        int bl2 = cdiv(tc, 256); if (bl2 > 2048) bl2 = 2048;
        reduce_ms<<<bl2, 256, 0, stream>>>(Li, stats + 4, NCLS, HW, P);
      }
      float* Ln = Lb[(prev+2)%3];
      long n = (long)NCLS*P;
      norm_add<<<cdiv(n,256), 256, 0, stream>>>(Lb[prev], Li, Ln, stats, NCLS, H, W, Hp, Wp);
      prev = (prev+2)%3;
    }
  }

  // ---- selection ----
  int HWf = Hh[7]*Wh[7], Pf = 2*HWf;
  pos_argmax<<<cdiv(Pf,256), 256, 0, stream>>>(Lb[prev], amax, mval, NCLS, Pf);
  select_kernel<<<2, 1024, 0, stream>>>(Lb[prev], amax, mval, (float*)d_out, NCLS, HWf);
}
